// MultiHeadAttention_11433202942205
// MI455X (gfx1250) — matrix operand order repaired and checked by reading, not compiled, not run on hardware
//
#include <hip/hip_runtime.h>

// ---------------------------------------------------------------------------
// MI455X (gfx1250) fused multi-head attention, bf16 WMMA pipeline.
// Memory-bound on the attn output (268MB) -> single-touch attn_bias/attn,
// scores kept in registers, all GEMMs via v_wmma_f32_16x16x32_bf16 (wave32).
// ---------------------------------------------------------------------------

typedef __bf16 v16bf __attribute__((ext_vector_type(16)));
typedef __bf16 v8bf  __attribute__((ext_vector_type(8)));
typedef float  v8f   __attribute__((ext_vector_type(8)));
typedef float  v4f   __attribute__((ext_vector_type(4)));

#define WMMA_BF16(a, b, c) \
  __builtin_amdgcn_wmma_f32_16x16x32_bf16(false, (a), false, (b), (short)0, (c), false, false)

constexpr int   Tq    = 1024;
constexpr int   Sk    = 1024;
constexpr int   Bb    = 4;
constexpr int   Ee    = 1024;
constexpr int   Hh    = 16;
constexpr int   Dd    = 64;
constexpr float SCALE = 0.125f;  // D^-0.5

// Dynamic LDS layout for attn_kernel (bytes)
constexpr int SM_QTILE = 0;                    // [16][64]  bf16  = 2048
constexpr int SM_PROBS = 2048;                 // [16][1024] bf16 = 32768
constexpr int SM_OPART = 2048 + 32768;         // [8][16][64] f32 = 32768
constexpr int SM_WSTAT = 2048 + 65536;         // [8][16] f32     = 512
constexpr int SM_RSTAT = 2048 + 65536 + 512;   // [16] f32        = 64
constexpr int ATTN_SMEM = SM_RSTAT + 64;       // 68160 B (< 320KB WGP LDS)

// ---- fragment loaders using contiguous K per lane half (same order in A and B) ----------
// A 16x32 bf16: lane m=lane&15; K = (lane>>4)*16 + e, contiguous
__device__ inline v16bf load_a_bf16(const __bf16* row, int k0, int lane) {
  const int kb = (lane >> 4) << 4;
  return *(const v16bf*)(row + k0 + kb);
}

__device__ inline v16bf load_a_f32(const float* row, int k0, int lane) {
  const int kb = (lane >> 4) << 4;
  const float* p = row + k0;
  v4f t0 = *(const v4f*)(p + kb);
  v4f t1 = *(const v4f*)(p + kb + 4);
  v4f t2 = *(const v4f*)(p + kb + 8);
  v4f t3 = *(const v4f*)(p + kb + 12);
  v16bf a;
#pragma unroll
  for (int i = 0; i < 4; ++i) {
    a[i]      = (__bf16)t0[i];
    a[i + 4]  = (__bf16)t1[i];
    a[i + 8]  = (__bf16)t2[i];
    a[i + 12] = (__bf16)t3[i];
  }
  return a;
}

// B 32x16 bf16: lane n=lane&15 is the column, K = (lane>>4)*16 + e, contiguous.
// Caller passes p already offset by (lane>>4)*16 within the column's K run.
__device__ inline v16bf load_b_f32(const float* p) {
  v16bf b;
#pragma unroll
  for (int i = 0; i < 16; i += 4) {
    v4f t = *(const v4f*)(p + i);
#pragma unroll
    for (int q = 0; q < 4; ++q) b[i + q] = (__bf16)t[q];
  }
  return b;
}

// ---------------------------------------------------------------------------
// Projection: Y(4096x1024) = X(4096x1024,f32) @ W^T(f32), stored bf16 per-head.
// Grid (8, 256): blockIdx.x = 128-col group, blockIdx.y = 16-row tile.
// 8 waves, each wave owns one 16x16 N-subtile; K loop 1024 in steps of 32.
// ---------------------------------------------------------------------------
template <bool TRANSPOSE_OUT>
__global__ __launch_bounds__(256) void proj_kernel(const float* __restrict__ X,
                                                   const float* __restrict__ W,
                                                   __bf16* __restrict__ Y, float scale) {
  const int lane = threadIdx.x & 31;
  const int wv   = threadIdx.x >> 5;
  const int n16  = lane & 15;
  const int hi   = lane >> 4;
  const int m0   = blockIdx.y << 4;
  const int j    = (blockIdx.x << 7) + (wv << 4) + n16;  // output column in E

  const float* arow = X + (size_t)(m0 + n16) * Ee;
  const float* brow = W + (size_t)j * Ee + (hi << 4);    // B col j = W row j

  v8f acc = {};
#pragma unroll 4
  for (int k0 = 0; k0 < Ee; k0 += 32) {
    v16bf a = load_a_f32(arow, k0, lane);
    v16bf b = load_b_f32(brow + k0);
    acc = WMMA_BF16(a, b, acc);
  }

  // C layout: VGPR r, lanes 0-15 -> M=r, lanes 16-31 -> M=r+8; N = lane&15.
  const int h = j >> 6, d = j & 63;
#pragma unroll
  for (int r = 0; r < 8; ++r) {
    const int m  = m0 + r + (hi << 3);
    const int t  = m >> 2;      // row index = t*B + b, B=4
    const int bb = m & 3;
    const int bh = bb * Hh + h;
    const __bf16 val = (__bf16)(acc[r] * scale);
    if (TRANSPOSE_OUT)
      Y[((size_t)(bh * Dd + d) << 10) + t] = val;        // (B*H, D, S)  (V^T)
    else
      Y[(((size_t)bh << 10) + t) * Dd + d] = val;        // (B*H, T|S, D)
  }
}

// ---------------------------------------------------------------------------
// Fused scores -> softmax -> attn write -> PV.  Grid (T/16, B*H), 256 thr.
// Wave w owns key columns [w*128, w*128+128); scores live in 8 v8f accums.
// ---------------------------------------------------------------------------
__global__ __launch_bounds__(256) void attn_kernel(const __bf16* __restrict__ qh,
                                                   const __bf16* __restrict__ kh,
                                                   const __bf16* __restrict__ vT,
                                                   const float* __restrict__ bias,
                                                   const unsigned char* __restrict__ mask,
                                                   float* __restrict__ attn_out,
                                                   __bf16* __restrict__ ctx) {
  extern __shared__ char smem[];
  __bf16* qtile = (__bf16*)(smem + SM_QTILE);
  __bf16* probs = (__bf16*)(smem + SM_PROBS);
  float*  opart = (float*)(smem + SM_OPART);
  float*  wstat = (float*)(smem + SM_WSTAT);
  float*  rstat = (float*)(smem + SM_RSTAT);

  const int tid  = threadIdx.x;
  const int lane = tid & 31;
  const int wv   = tid >> 5;
  const int n16  = lane & 15;
  const int hi   = lane >> 4;

  const int bh = blockIdx.y;
  const int b  = bh >> 4;
  const int h  = bh & 15;
  const int t0 = blockIdx.x << 4;

  // Stage the 16x64 Q tile (bf16, pre-scaled by proj) to LDS: 8B per thread.
  {
    const __bf16* src = qh + (((size_t)bh << 10) + t0) * Dd;
    ((uint2*)qtile)[tid] = ((const uint2*)src)[tid];
  }
  __syncthreads();

  const int jw = wv << 7;  // this wave's key-column slice start
  const v16bf qa0 = load_a_bf16(qtile + n16 * Dd, 0, lane);
  const v16bf qa1 = load_a_bf16(qtile + n16 * Dd, 32, lane);

  v8f zero = {};
  v8f c[8];
#pragma unroll
  for (int i = 0; i < 8; ++i) c[i] = zero;

  // scores = Q . K^T  (B-frag rows of kh are contiguous in d)
#pragma unroll
  for (int jt = 0; jt < 8; ++jt) {
    const __bf16* kp =
        kh + (((size_t)bh << 10) + (jw + (jt << 4) + n16)) * Dd + (hi << 4);
    v16bf b0 = *(const v16bf*)kp;         // K = d in [0,32)
    v16bf b1 = *(const v16bf*)(kp + 32);  // K = d in [32,64)
    c[jt] = WMMA_BF16(qa0, b0, c[jt]);
    c[jt] = WMMA_BF16(qa1, b1, c[jt]);
  }

  // + attn_bias; row max over raw (pre-mask) scores, as in the reference
  const float* bias_base = bias + ((size_t)bh << 20) + ((size_t)t0 << 10);
  float rmax[8];
#pragma unroll
  for (int r = 0; r < 8; ++r) rmax[r] = -__builtin_inff();
#pragma unroll
  for (int jt = 0; jt < 8; ++jt) {
    const int jcol = jw + (jt << 4) + n16;
#pragma unroll
    for (int r = 0; r < 8; ++r) {
      const int row = (hi << 3) + r;
      float s = c[jt][r] + bias_base[((size_t)row << 10) + jcol];
      c[jt][r] = s;
      rmax[r] = fmaxf(rmax[r], s);
    }
  }
#pragma unroll
  for (int msk = 8; msk >= 1; msk >>= 1)
#pragma unroll
    for (int r = 0; r < 8; ++r) rmax[r] = fmaxf(rmax[r], __shfl_xor(rmax[r], msk, 32));
  if (n16 == 0) {
#pragma unroll
    for (int r = 0; r < 8; ++r) wstat[wv * 16 + (hi << 3) + r] = rmax[r];
  }
  __syncthreads();
  if (tid < 16) {
    float m = wstat[tid];
#pragma unroll
    for (int w = 1; w < 8; ++w) m = fmaxf(m, wstat[w * 16 + tid]);
    rstat[tid] = m;
  }
  __syncthreads();
#pragma unroll
  for (int r = 0; r < 8; ++r) rmax[r] = rstat[(hi << 3) + r];

  // exp(s - rowmax), key-padding mask -> 0; accumulate row sums
  float rsum[8];
#pragma unroll
  for (int r = 0; r < 8; ++r) rsum[r] = 0.f;
#pragma unroll
  for (int jt = 0; jt < 8; ++jt) {
    const int jcol = jw + (jt << 4) + n16;
    const bool dead = mask[(b << 10) + jcol] != 0;
#pragma unroll
    for (int r = 0; r < 8; ++r) {
      float p = dead ? 0.f : __expf(c[jt][r] - rmax[r]);
      c[jt][r] = p;
      rsum[r] += p;
    }
  }
#pragma unroll
  for (int msk = 8; msk >= 1; msk >>= 1)
#pragma unroll
    for (int r = 0; r < 8; ++r) rsum[r] += __shfl_xor(rsum[r], msk, 32);
  __syncthreads();  // all rstat(max)/wstat reads done before reuse
  if (n16 == 0) {
#pragma unroll
    for (int r = 0; r < 8; ++r) wstat[wv * 16 + (hi << 3) + r] = rsum[r];
  }
  __syncthreads();
  if (tid < 16) {
    float s = wstat[tid];
#pragma unroll
    for (int w = 1; w < 8; ++w) s += wstat[w * 16 + tid];
    rstat[tid] = s;
  }
  __syncthreads();
  float rinv[8];
#pragma unroll
  for (int r = 0; r < 8; ++r) rinv[r] = 1.0f / rstat[(hi << 3) + r];

  // Write normalized attn (fp32, single touch of the 268MB tensor) + bf16 probs
  float* aout = attn_out + ((size_t)bh << 20) + ((size_t)t0 << 10);
#pragma unroll
  for (int jt = 0; jt < 8; ++jt) {
    const int jcol = jw + (jt << 4) + n16;
#pragma unroll
    for (int r = 0; r < 8; ++r) {
      const int row = (hi << 3) + r;
      const float p = c[jt][r] * rinv[r];
      aout[((size_t)row << 10) + jcol] = p;
      probs[(row << 10) + jcol] = (__bf16)p;
    }
  }
  __syncthreads();

  // PV split-K: each wave contributes probs[:, jw:jw+128] . V^T -> 16x64 partial
  v8f oc[4];
#pragma unroll
  for (int i = 0; i < 4; ++i) oc[i] = zero;
  const __bf16* vb = vT + (((size_t)bh * Dd) << 10);
#pragma unroll
  for (int ks = 0; ks < 4; ++ks) {
    const int k0 = jw + (ks << 5);
    v16bf pa = load_a_bf16(probs + (n16 << 10), k0, lane);
#pragma unroll
    for (int nt = 0; nt < 4; ++nt) {
      const int d = (nt << 4) + n16;
      v16bf bv = *(const v16bf*)(vb + ((size_t)d << 10) + k0 + (hi << 4));  // V^T row d
      oc[nt] = WMMA_BF16(pa, bv, oc[nt]);
    }
  }
#pragma unroll
  for (int nt = 0; nt < 4; ++nt)
#pragma unroll
    for (int r = 0; r < 8; ++r)
      opart[((wv << 4) + (hi << 3) + r) * 64 + (nt << 4) + n16] = oc[nt][r];
  __syncthreads();

  // Cross-wave reduction; ctx stored bf16 as (T*B, E) with row = t*B + b
  {
    const int m  = tid >> 4;
    const int d0 = (tid & 15) << 2;
    v4f s = *(const v4f*)(opart + (m * 64 + d0));
#pragma unroll
    for (int w = 1; w < 8; ++w) s += *(const v4f*)(opart + ((w * 16 + m) * 64 + d0));
    __bf16* cp = ctx + (((size_t)((t0 + m) * Bb + b)) << 10) + (h << 6) + d0;
    cp[0] = (__bf16)s[0]; cp[1] = (__bf16)s[1]; cp[2] = (__bf16)s[2]; cp[3] = (__bf16)s[3];
  }
}

// ---------------------------------------------------------------------------
// Output projection: out(4096x1024,f32) = ctx(bf16) @ Wp^T + bp
// ---------------------------------------------------------------------------
__global__ __launch_bounds__(256) void outproj_kernel(const __bf16* __restrict__ ctx,
                                                      const float* __restrict__ Wp,
                                                      const float* __restrict__ bp,
                                                      float* __restrict__ out) {
  const int lane = threadIdx.x & 31;
  const int wv   = threadIdx.x >> 5;
  const int n16  = lane & 15;
  const int hi   = lane >> 4;
  const int m0   = blockIdx.y << 4;
  const int j    = (blockIdx.x << 7) + (wv << 4) + n16;

  const __bf16* arow = ctx + ((size_t)(m0 + n16) << 10);
  const float*  brow = Wp + (size_t)j * Ee + (hi << 4);

  v8f acc = {};
#pragma unroll 4
  for (int k0 = 0; k0 < Ee; k0 += 32) {
    v16bf a = load_a_bf16(arow, k0, lane);
    v16bf b = load_b_f32(brow + k0);
    acc = WMMA_BF16(a, b, acc);
  }

  const float bias = bp[j];
#pragma unroll
  for (int r = 0; r < 8; ++r)
    out[((size_t)(m0 + r + (hi << 3)) << 10) + j] = acc[r] + bias;
}

// ---------------------------------------------------------------------------
extern "C" void kernel_launch(void* const* d_in, const int* in_sizes, int n_in,
                              void* d_out, int out_size, void* d_ws, size_t ws_size,
                              hipStream_t stream) {
  const float* q  = (const float*)d_in[0];
  const float* k  = (const float*)d_in[1];
  const float* v  = (const float*)d_in[2];
  const float* Wq = (const float*)d_in[3];
  const float* Wk = (const float*)d_in[4];
  const float* Wv = (const float*)d_in[5];
  const float* Wp = (const float*)d_in[6];
  const float* bp = (const float*)d_in[7];
  const float* attn_bias = (const float*)d_in[8];
  const unsigned char* mask = (const unsigned char*)d_in[9];

  float* out_proj = (float*)d_out;                              // (T,B,E)
  float* attn_out = (float*)d_out + (size_t)Tq * Bb * Ee;       // (B,H,T,S)

  __bf16* wsb = (__bf16*)d_ws;
  const size_t headElems = (size_t)Bb * Hh * 1024 * Dd;  // 4,194,304 per tensor
  __bf16* qh  = wsb;                  // (B*H, T, D)  pre-scaled
  __bf16* kh  = wsb + headElems;      // (B*H, S, D)
  __bf16* vTt = wsb + 2 * headElems;  // (B*H, D, S)  transposed
  __bf16* ctx = wsb + 3 * headElems;  // (T*B, E)

  dim3 blk(256);
  dim3 gproj(8, 256);
  proj_kernel<false><<<gproj, blk, 0, stream>>>(q, Wq, qh, SCALE);
  proj_kernel<false><<<gproj, blk, 0, stream>>>(k, Wk, kh, 1.0f);
  proj_kernel<true ><<<gproj, blk, 0, stream>>>(v, Wv, vTt, 1.0f);

  dim3 gattn(Tq / 16, Bb * Hh);
  attn_kernel<<<gattn, blk, ATTN_SMEM, stream>>>(qh, kh, vTt, attn_bias, mask,
                                                 attn_out, ctx);

  outproj_kernel<<<gproj, blk, 0, stream>>>(ctx, Wp, bp, out_proj);
}
